// TimeSeriesForecaster_2319282340624
// MI455X (gfx1250) — matrix operand order repaired and checked by reading, not compiled, not run on hardware
//
#include <hip/hip_runtime.h>
#include <hip/hip_bf16.h>
#include <math.h>

// ---------------- problem constants (from reference) ----------------
#define BATCH   8
#define LSEQ    8192
#define ENC_IN  21
#define D_MODEL 512
#define D_STATE 16
#define D_CONV  4
#define D_INNER 1024
#define DT_RANK 32
#define PRED    720
#define C_OUT   21

typedef __attribute__((ext_vector_type(16))) __bf16 v16bf;
typedef __attribute__((ext_vector_type(8)))  __bf16 v8bf;
typedef __attribute__((ext_vector_type(8)))  float  v8f;

struct bf16x16 { v8bf lo; v8bf hi; };

__device__ __forceinline__ v16bf load_frag16(const __bf16* p) {
    v8bf lo = *(const v8bf*)(p);
    v8bf hi = *(const v8bf*)(p + 8);
    bf16x16 pr{lo, hi};
    return __builtin_bit_cast(v16bf, pr);
}

// ---------------- stats: mean / stdev per (b, channel) ----------------
__global__ void __launch_bounds__(256) stats_kernel(const float* __restrict__ x,
                                                    float* __restrict__ mean,
                                                    float* __restrict__ stdev) {
    int bc = blockIdx.x;               // b*ENC_IN + c
    int b = bc / ENC_IN, c = bc % ENC_IN;
    __shared__ float s1[256], s2[256];
    float a = 0.f, q = 0.f;
    for (int l = threadIdx.x; l < LSEQ; l += 256) {
        float v = x[((size_t)b * LSEQ + l) * ENC_IN + c];
        a += v; q += v * v;
    }
    s1[threadIdx.x] = a; s2[threadIdx.x] = q;
    __syncthreads();
    for (int s = 128; s > 0; s >>= 1) {
        if (threadIdx.x < s) { s1[threadIdx.x] += s1[threadIdx.x + s]; s2[threadIdx.x] += s2[threadIdx.x + s]; }
        __syncthreads();
    }
    if (threadIdx.x == 0) {
        float mu  = s1[0] / (float)LSEQ;
        float var = s2[0] / (float)LSEQ - mu * mu;
        mean[bc]  = mu;
        stdev[bc] = sqrtf(var + 1e-5f);
    }
}

// ---------------- embed: ((x-mu)/sd) @ W_embed + b  ->  bf16 ----------------
__global__ void __launch_bounds__(512) embed_kernel(const float* __restrict__ x,
                                                    const float* __restrict__ mean,
                                                    const float* __restrict__ stdev,
                                                    const float* __restrict__ We,
                                                    const float* __restrict__ be,
                                                    __bf16* __restrict__ emb) {
    int row = blockIdx.x;              // b*LSEQ + l
    int b = row / LSEQ;
    __shared__ float xs[ENC_IN];
    if (threadIdx.x < ENC_IN) {
        float v = x[(size_t)row * ENC_IN + threadIdx.x];
        xs[threadIdx.x] = (v - mean[b * ENC_IN + threadIdx.x]) / stdev[b * ENC_IN + threadIdx.x];
    }
    __syncthreads();
    int j = threadIdx.x;               // 0..511
    float s = be[j];
    #pragma unroll
    for (int c = 0; c < ENC_IN; ++c) s += xs[c] * We[c * D_MODEL + j];
    emb[(size_t)row * D_MODEL + j] = (__bf16)s;
}

// ---------------- pack f32 B[K,N] (row-major, ldb) into WMMA fragment layout ----
// packed index: ((nTile*kTiles + kTile)*32 + lane)*16 + i, lane={half,n},
// elem = B[kTile*32 + half*16 + i][nTile*16 + n]; per-lane 16 bf16 contiguous (32 B)
__global__ void pack_b_kernel(const float* __restrict__ src, __bf16* __restrict__ dst,
                              int K, int N, int ldb) {
    int kTiles = K >> 5, nTiles = N >> 4;
    int idx = blockIdx.x * blockDim.x + threadIdx.x;
    if (idx >= kTiles * nTiles * 32) return;
    int lane = idx & 31; int t = idx >> 5;
    int kTile = t % kTiles, nTile = t / kTiles;
    int n = lane & 15, half = lane >> 4;
    const float* s = src + (size_t)(kTile * 32 + half * 16) * ldb + nTile * 16 + n;
    __bf16* d = dst + (((size_t)nTile * kTiles + kTile) * 32 + lane) * 16;
    #pragma unroll
    for (int i = 0; i < 16; ++i) d[i] = (__bf16)s[(size_t)i * ldb];
}

// ---------------- strided extract f32 -> bf16 (dt columns of x_dbl) ----------------
__global__ void extract_bf16_kernel(const float* __restrict__ src, __bf16* __restrict__ dst,
                                    long rows, int cols, int src_ld) {
    long i = (long)blockIdx.x * blockDim.x + threadIdx.x;
    if (i >= rows * (long)cols) return;
    long r = i / cols; int c = (int)(i % cols);
    dst[i] = (__bf16)src[r * src_ld + c];
}

// ---------------- WMMA bf16 GEMM: C = A[M,K] * B(packed)  (+bias, epilogue) ------
// block = (32,8): 8 waves, each computes a 32x64 tile (2 row frags x 4 col sub-tiles).
// Each B fragment (2x b128, L2-resident) feeds two independent back-to-back WMMAs.
// epilogue: 0 = store f32, 1 = store bf16, 2 = softplus -> bf16
__global__ void __launch_bounds__(256) gemm_bf16_wmma(const __bf16* __restrict__ A,
                                                      const __bf16* __restrict__ pB,
                                                      float*  __restrict__ Cf,
                                                      __bf16* __restrict__ Cb,
                                                      const float* __restrict__ bias,
                                                      int M, int lda, int ldc,
                                                      int kTiles, int nTileBase,
                                                      int epilogue) {
    const int lane = threadIdx.x;            // 0..31
    const int wave = threadIdx.y;            // 0..7
    const int row0 = (blockIdx.y * 8 + wave) * 32;
    const int nt0  = blockIdx.x * 4;
    const int m    = lane & 15;
    const int half = lane >> 4;

    v8f z8 = {0.f,0.f,0.f,0.f,0.f,0.f,0.f,0.f};
    v8f acc0[4] = {z8, z8, z8, z8};
    v8f acc1[4] = {z8, z8, z8, z8};

    const __bf16* aRow0 = A + (size_t)(row0 + m) * lda + half * 16;
    const __bf16* aRow1 = aRow0 + (size_t)16 * lda;

    for (int kt = 0; kt < kTiles; ++kt) {
        // two A fragments (rows row0..+16 and row0+16..+32)
        v16bf a0 = load_frag16(aRow0 + kt * 32);
        v16bf a1 = load_frag16(aRow1 + kt * 32);

        if (kt + 1 < kTiles) {
            __builtin_prefetch(aRow0 + kt * 32 + 32, 0, 3);
            __builtin_prefetch(aRow1 + kt * 32 + 32, 0, 3);
        }

        #pragma unroll
        for (int t = 0; t < 4; ++t) {
            v16bf b = load_frag16(pB +
                (((size_t)(nTileBase + nt0 + t) * kTiles + kt) * 32 + lane) * 16);
            acc0[t] = __builtin_amdgcn_wmma_f32_16x16x32_bf16(
                          false, a0, false, b, (short)0, acc0[t], false, false);
            acc1[t] = __builtin_amdgcn_wmma_f32_16x16x32_bf16(
                          false, a1, false, b, (short)0, acc1[t], false, false);
        }
    }

    // epilogue: D layout — VGPR v: row = v + half*8, col = lane&15 within sub-tile
    #pragma unroll
    for (int t = 0; t < 4; ++t) {
        #pragma unroll
        for (int v = 0; v < 8; ++v) {
            int c  = (nt0 + t) * 16 + m;
            float bi = bias ? bias[c] : 0.f;
            #pragma unroll
            for (int g = 0; g < 2; ++g) {
                int r = row0 + g * 16 + half * 8 + v;
                if (r >= M) continue;
                float f = (g == 0 ? acc0[t][v] : acc1[t][v]) + bi;
                if (epilogue == 2) f = (f > 20.f) ? f : log1pf(__expf(f));
                if (epilogue == 0) Cf[(size_t)r * ldc + c] = f;
                else               Cb[(size_t)r * ldc + c] = (__bf16)f;
            }
        }
    }
}

// ---------------- depthwise causal conv(4) + bias + SiLU ----------------
__global__ void conv_silu_kernel(const __bf16* __restrict__ u,
                                 const float* __restrict__ cw,
                                 const float* __restrict__ cb,
                                 __bf16* __restrict__ up) {
    size_t idx = (size_t)blockIdx.x * blockDim.x + threadIdx.x;
    if (idx >= (size_t)BATCH * LSEQ * D_INNER) return;
    int d = (int)(idx % D_INNER);
    size_t rl = idx / D_INNER;
    int l = (int)(rl % LSEQ);
    size_t b = rl / LSEQ;
    float s = cb[d];
    #pragma unroll
    for (int t = 0; t < D_CONV; ++t) {
        int ll = l - (D_CONV - 1) + t;
        if (ll >= 0) s += (float)u[(b * LSEQ + ll) * D_INNER + d] * cw[d * D_CONV + t];
    }
    float sig = 1.f / (1.f + __expf(-s));
    up[idx] = (__bf16)(s * sig);
}

// ---------------- selective scan (sequential over L), fused gating ----------------
__global__ void __launch_bounds__(1024) scan_kernel(const float* __restrict__ xdbl,
                                                    const __bf16* __restrict__ delta,
                                                    const __bf16* __restrict__ up,
                                                    const __bf16* __restrict__ zb,
                                                    const float* __restrict__ A_log,
                                                    const float* __restrict__ Dvec,
                                                    __bf16* __restrict__ yg) {
    int b = blockIdx.x;
    int d = threadIdx.x;
    __shared__ float Bsh[D_STATE], Csh[D_STATE];
    float As[D_STATE], h[D_STATE];
    #pragma unroll
    for (int s = 0; s < D_STATE; ++s) { As[s] = -__expf(A_log[d * D_STATE + s]); h[s] = 0.f; }
    float Dd = Dvec[d];

    for (int l = 0; l < LSEQ; ++l) {
        size_t row = (size_t)b * LSEQ + l;
        if (d < D_STATE)            Bsh[d]           = xdbl[row * 64 + DT_RANK + d];
        else if (d < 2 * D_STATE)   Csh[d - D_STATE] = xdbl[row * 64 + DT_RANK + D_STATE + (d - D_STATE)];
        __syncthreads();
        float dt = (float)delta[row * D_INNER + d];
        float uu = (float)up[row * D_INNER + d];
        float du = dt * uu;
        float y = 0.f;
        #pragma unroll
        for (int s = 0; s < D_STATE; ++s) {
            float dA = __expf(dt * As[s]);
            float hh = dA * h[s] + du * Bsh[s];
            h[s] = hh;
            y += hh * Csh[s];
        }
        if (l >= LSEQ - PRED) {
            float z = (float)zb[row * D_INNER + d];
            float g = z / (1.f + __expf(-z));
            float outv = (y + uu * Dd) * g;
            yg[((size_t)b * PRED + (l - (LSEQ - PRED))) * D_INNER + d] = (__bf16)outv;
        }
        __syncthreads();
    }
}

// ---------------- final projection (N=21) + de-normalization ----------------
__global__ void __launch_bounds__(32) final_proj_kernel(const float* __restrict__ yo,
                                                        const float* __restrict__ Wp,
                                                        const float* __restrict__ bp,
                                                        const float* __restrict__ mean,
                                                        const float* __restrict__ stdev,
                                                        float* __restrict__ out) {
    int row = blockIdx.x;                 // b*PRED + t
    int b = row / PRED;
    int c = threadIdx.x;
    if (c >= C_OUT) return;
    float s = bp[c];
    const float* yr = yo + (size_t)row * D_MODEL;
    for (int k = 0; k < D_MODEL; ++k) s += yr[k] * Wp[k * C_OUT + c];
    out[(size_t)row * C_OUT + c] = s * stdev[b * C_OUT + c] + mean[b * C_OUT + c];
}

// =====================================================================
extern "C" void kernel_launch(void* const* d_in, const int* in_sizes, int n_in,
                              void* d_out, int out_size, void* d_ws, size_t ws_size,
                              hipStream_t stream) {
    const float* x_enc     = (const float*)d_in[0];
    const float* W_embed   = (const float*)d_in[4];
    const float* b_embed   = (const float*)d_in[5];
    const float* W_in      = (const float*)d_in[6];
    const float* conv_w    = (const float*)d_in[7];
    const float* conv_b    = (const float*)d_in[8];
    const float* W_xproj   = (const float*)d_in[9];
    const float* W_dt      = (const float*)d_in[10];
    const float* b_dt      = (const float*)d_in[11];
    const float* A_log     = (const float*)d_in[12];
    const float* Dvec      = (const float*)d_in[13];
    const float* W_out_ssm = (const float*)d_in[14];
    const float* W_proj    = (const float*)d_in[15];
    const float* b_proj    = (const float*)d_in[16];
    float* out = (float*)d_out;

    const size_t ROWS      = (size_t)BATCH * LSEQ;     // 65536
    const size_t PROWS     = (size_t)BATCH * PRED;     // 5760
    const size_t PROWS_PAD = 5888;                     // pad to multiple of 256

    // ---- workspace layout ----
    size_t off = 0;
    auto alloc = [&](size_t bytes) -> char* {
        char* p = (char*)d_ws + off;
        off += (bytes + 255) & ~(size_t)255;
        return p;
    };
    float*  mean_   = (float*) alloc(BATCH * ENC_IN * 4);
    float*  stdev_  = (float*) alloc(BATCH * ENC_IN * 4);
    __bf16* emb     = (__bf16*)alloc(ROWS * D_MODEL * 2);                 // 64 MB
    __bf16* pWin    = (__bf16*)alloc((size_t)D_MODEL * 2 * D_INNER * 2);  // 2 MB (packed)
    __bf16* pWxp    = (__bf16*)alloc((size_t)D_INNER * 64 * 2);
    __bf16* pWdt    = (__bf16*)alloc((size_t)DT_RANK * D_INNER * 2);
    __bf16* pWout   = (__bf16*)alloc((size_t)D_INNER * D_MODEL * 2);
    __bf16* u_raw   = (__bf16*)alloc(ROWS * D_INNER * 2);                 // 128 MB (reused as delta)
    __bf16* z_b     = (__bf16*)alloc(ROWS * D_INNER * 2);                 // 128 MB
    __bf16* u_post  = (__bf16*)alloc(ROWS * D_INNER * 2);                 // 128 MB
    float*  x_dbl   = (float*) alloc(ROWS * 64 * 4);                      // 16 MB
    __bf16* dt_b    = (__bf16*)alloc(ROWS * DT_RANK * 2);                 // 4 MB
    __bf16* y_gated = (__bf16*)alloc(PROWS_PAD * D_INNER * 2);            // 12 MB (padded)
    float*  y_out   = (float*) alloc(PROWS * D_MODEL * 4);                // 12 MB
    __bf16* delta_b = u_raw;   // alias: u_raw dead after conv, delta produced after

    // 1) per-(b,c) mean / stdev
    stats_kernel<<<BATCH * ENC_IN, 256, 0, stream>>>(x_enc, mean_, stdev_);

    // 2) normalize + embed -> bf16
    embed_kernel<<<(int)ROWS, D_MODEL, 0, stream>>>(x_enc, mean_, stdev_, W_embed, b_embed, emb);

    // 3) pack weights (f32 -> bf16 fragment layout)
    auto pack = [&](const float* s, __bf16* d, int K, int N, int ldb) {
        int total = (K >> 5) * (N >> 4) * 32;
        pack_b_kernel<<<(total + 255) / 256, 256, 0, stream>>>(s, d, K, N, ldb);
    };
    pack(W_in,      pWin,  D_MODEL, 2 * D_INNER, 2 * D_INNER);
    pack(W_xproj,   pWxp,  D_INNER, 64,          64);
    pack(W_dt,      pWdt,  DT_RANK, D_INNER,     D_INNER);
    pack(W_out_ssm, pWout, D_INNER, D_MODEL,     D_MODEL);

    dim3 blk(32, 8);
    // 4) u_raw = emb @ W_in[:, :1024]   (M=65536, K=512, N=1024) -> bf16
    gemm_bf16_wmma<<<dim3(D_INNER / 64, (int)(ROWS / 256)), blk, 0, stream>>>(
        emb, pWin, nullptr, u_raw, nullptr,
        (int)ROWS, D_MODEL, D_INNER, D_MODEL / 32, /*nTileBase=*/0, 1);
    // 5) z = emb @ W_in[:, 1024:]  (nTile offset 64 into packed W_in)
    gemm_bf16_wmma<<<dim3(D_INNER / 64, (int)(ROWS / 256)), blk, 0, stream>>>(
        emb, pWin, nullptr, z_b, nullptr,
        (int)ROWS, D_MODEL, D_INNER, D_MODEL / 32, /*nTileBase=*/D_INNER / 16, 1);

    // 6) causal depthwise conv + SiLU
    {
        size_t total = ROWS * D_INNER;
        conv_silu_kernel<<<(int)((total + 255) / 256), 256, 0, stream>>>(u_raw, conv_w, conv_b, u_post);
    }

    // 7) x_dbl = u_post @ W_xproj  (M=65536, K=1024, N=64) -> f32
    gemm_bf16_wmma<<<dim3(1, (int)(ROWS / 256)), blk, 0, stream>>>(
        u_post, pWxp, x_dbl, nullptr, nullptr,
        (int)ROWS, D_INNER, 64, D_INNER / 32, 0, 0);

    // 8) extract dt columns (0..31) of x_dbl as bf16
    {
        long n = (long)ROWS * DT_RANK;
        extract_bf16_kernel<<<(int)((n + 255) / 256), 256, 0, stream>>>(x_dbl, dt_b, (long)ROWS, DT_RANK, 64);
    }

    // 9) delta = softplus(dt @ W_dt + b_dt)  (M=65536, K=32, N=1024) -> bf16 (aliases u_raw)
    gemm_bf16_wmma<<<dim3(D_INNER / 64, (int)(ROWS / 256)), blk, 0, stream>>>(
        dt_b, pWdt, nullptr, delta_b, b_dt,
        (int)ROWS, DT_RANK, D_INNER, 1, 0, 2);

    // 10) selective scan with fused D*u + SiLU(z) gating, last PRED rows only
    scan_kernel<<<BATCH, D_INNER, 0, stream>>>(x_dbl, delta_b, u_post, z_b, A_log, Dvec, y_gated);

    // 11) y_out = y_gated @ W_out_ssm  (M=5760 real, 5888 padded, K=1024, N=512) -> f32
    gemm_bf16_wmma<<<dim3(D_MODEL / 64, (int)(PROWS_PAD / 256)), blk, 0, stream>>>(
        y_gated, pWout, y_out, nullptr, nullptr,
        (int)PROWS, D_INNER, D_MODEL, D_INNER / 32, 0, 0);

    // 12) final projection + de-normalization
    final_proj_kernel<<<(int)PROWS, 32, 0, stream>>>(y_out, W_proj, b_proj, mean_, stdev_, out);
}
